// GQA_49340584296502
// MI455X (gfx1250) — matrix operand order repaired and checked by reading, not compiled, not run on hardware
//
#include <hip/hip_runtime.h>
#include <stdint.h>

// ---------------------------------------------------------------------------
// Types for WMMA (CDNA5 / gfx1250, wave32)
// ---------------------------------------------------------------------------
typedef __attribute__((ext_vector_type(16))) __bf16        v16bf;
typedef __attribute__((ext_vector_type(8)))  float         v8f;
typedef __attribute__((ext_vector_type(4)))  unsigned int  v4u;
typedef __attribute__((ext_vector_type(4)))  int           v4i;

#define GLOBAL_AS __attribute__((address_space(1)))
#define LDS_AS    __attribute__((address_space(3)))

union Frag16 { v16bf v; v4u u[2]; };           // 16 bf16 = 8 VGPRs
union U4S8   { v4u u; unsigned short s[8]; };  // for LDS transpose scatter

// CDNA5 async global->LDS copy (ASYNCcnt-tracked), guarded so the file
// compiles on toolchains where the builtins are absent.
#if defined(__HIP_DEVICE_COMPILE__) &&                                        \
    __has_builtin(__builtin_amdgcn_global_load_async_to_lds_b128) &&          \
    __has_builtin(__builtin_amdgcn_s_wait_asynccnt)
#define ASYNC_OK 1
#else
#define ASYNC_OK 0
#endif

__device__ __forceinline__ unsigned short f2bf(float f) {
  union { float f; unsigned int u; } c; c.f = f;
  unsigned int u = c.u;
  u += 0x7FFFu + ((u >> 16) & 1u);   // round-to-nearest-even
  return (unsigned short)(u >> 16);
}

__device__ __forceinline__ void store_out(float* p, float v) { *p = v; }
__device__ __forceinline__ void store_out(unsigned short* p, float v) { *p = f2bf(v); }

// ---------------------------------------------------------------------------
// fp32 -> bf16 conversion (plain, for x)
// ---------------------------------------------------------------------------
__global__ void f32_to_bf16_kernel(const float* __restrict__ in,
                                   unsigned short* __restrict__ out, size_t n) {
  size_t i = (size_t)blockIdx.x * blockDim.x + threadIdx.x;
  if (i < n) out[i] = f2bf(in[i]);
}

// ---------------------------------------------------------------------------
// fp32 [K,N] -> bf16 transposed [N,K] (for weights, so the GEMM B operand is
// already in WMMA B-fragment-friendly [n][k] order; coalesced both sides via
// a 32x32 LDS tile). One-time cost, bandwidth trivial at 23.3 TB/s.
// ---------------------------------------------------------------------------
__global__ __launch_bounds__(256)
void f32_to_bf16_transpose_kernel(const float* __restrict__ in,   // [K,N]
                                  unsigned short* __restrict__ out, // [N,K]
                                  int K, int N) {
  __shared__ float tile[32][33];
  const int n0 = blockIdx.x * 32;
  const int k0 = blockIdx.y * 32;
  const int tx = threadIdx.x & 31;
  const int ty = threadIdx.x >> 5;           // 0..7
#pragma unroll
  for (int i = 0; i < 32; i += 8)
    tile[ty + i][tx] = in[(size_t)(k0 + ty + i) * N + n0 + tx];
  __syncthreads();
#pragma unroll
  for (int i = 0; i < 32; i += 8)
    out[(size_t)(n0 + ty + i) * K + k0 + tx] = f2bf(tile[tx][ty + i]);
}

// ---------------------------------------------------------------------------
// bf16 WMMA GEMM: C[M,N] = A[M,K] * Bt[N,K]^T + bias
// 128 threads = 4 wave32; block tile 128x128, K-step 64.
// Wave grid 2(M) x 2(N); each wave computes 64x64 via 4x4 16x16 accumulators
// -> 16 WMMA per 16 LDS b128 fragment loads per k32 step.
// Double-buffered LDS staged with GLOBAL_LOAD_ASYNC_TO_LDS_B128 when
// available (ASYNCcnt + barrier pipeline), else sync copy fallback.
// ---------------------------------------------------------------------------
template <typename OutT>
__global__ __launch_bounds__(128)
void gemm_bf16_wmma(const unsigned short* __restrict__ A,   // [M,K]
                    const unsigned short* __restrict__ Bt,  // [N,K]
                    const float* __restrict__ bias,
                    OutT* __restrict__ C,
                    int M, int N, int K)
{
  __shared__ __align__(16) unsigned short lA[2][128][72];  // [m][k], pad 8
  __shared__ __align__(16) unsigned short lB[2][128][72];  // [n][k], pad 8

  const int tid   = threadIdx.x;
  const int lane  = tid & 31;
  const int wid   = tid >> 5;
  const int wm    = wid >> 1;      // 0..1 (64 rows)
  const int wn    = wid & 1;       // 0..1 (64 cols)
  const int m0    = blockIdx.y * 128;
  const int n0    = blockIdx.x * 128;
  const int mlane = lane & 15;
  const int half  = lane >> 4;

  v8f acc[4][4];
  const v8f vz = {0.f,0.f,0.f,0.f,0.f,0.f,0.f,0.f};
#pragma unroll
  for (int i = 0; i < 4; ++i)
#pragma unroll
    for (int j = 0; j < 4; ++j) acc[i][j] = vz;

  // Stage one 128x64 A tile + 128x64 B tile into buffer `buf`.
  auto copy_tile = [&](int buf, int k0) {
#pragma unroll
    for (int i = 0; i < 8; ++i) {
      int v = tid + i * 128;
      int r = v >> 3, c = (v & 7) * 8;      // r: 0..127, c: 0..56
      const unsigned short* ga = A  + (size_t)(m0 + r) * K + k0 + c;
      const unsigned short* gb = Bt + (size_t)(n0 + r) * K + k0 + c;
#if ASYNC_OK
      __builtin_amdgcn_global_load_async_to_lds_b128(
          (GLOBAL_AS v4i*)ga, (LDS_AS v4i*)&lA[buf][r][c], 0, 0);
      __builtin_amdgcn_global_load_async_to_lds_b128(
          (GLOBAL_AS v4i*)gb, (LDS_AS v4i*)&lB[buf][r][c], 0, 0);
#else
      *(v4u*)&lA[buf][r][c] = *(const v4u*)ga;
      *(v4u*)&lB[buf][r][c] = *(const v4u*)gb;
#endif
    }
  };

  auto compute = [&](int buf) {
#pragma unroll
    for (int kk = 0; kk < 64; kk += 32) {
      // A fragments: 16x32 bf16, k = (v/4)*16 + half*8 + (v%4)*2
      Frag16 af[4];
#pragma unroll
      for (int i = 0; i < 4; ++i) {
        const unsigned short* p = &lA[buf][wm * 64 + i * 16 + mlane][kk + half * 8];
        af[i].u[0] = *(const v4u*)p;
        af[i].u[1] = *(const v4u*)(p + 16);
      }
      // B fragments: 32x16 bf16, k = (v/4)*16 + half*8 + (v%4)*2
      Frag16 bfg[4];
#pragma unroll
      for (int j = 0; j < 4; ++j) {
        const unsigned short* p = &lB[buf][wn * 64 + j * 16 + mlane][kk + half * 8];
        bfg[j].u[0] = *(const v4u*)p;
        bfg[j].u[1] = *(const v4u*)(p + 16);
      }
#pragma unroll
      for (int i = 0; i < 4; ++i)
#pragma unroll
        for (int j = 0; j < 4; ++j)
          acc[i][j] = __builtin_amdgcn_wmma_f32_16x16x32_bf16(
              false, af[i].v, false, bfg[j].v, (short)0, acc[i][j], false, false);
    }
  };

  // ---- software pipeline: prefetch next tile while computing current ----
  copy_tile(0, 0);
#if ASYNC_OK
  __builtin_amdgcn_s_wait_asynccnt(0);
#endif
  __syncthreads();

  int cur = 0;
  for (int k0 = 0; k0 < K; k0 += 64) {
    const int nxt = k0 + 64;
    if (nxt < K) copy_tile(cur ^ 1, nxt);   // async: overlaps with WMMA below
    compute(cur);
#if ASYNC_OK
    if (nxt < K) __builtin_amdgcn_s_wait_asynccnt(0);
#endif
    __syncthreads();   // all waves done reading `cur` + next buffer visible
    cur ^= 1;
  }

  // ---- epilogue: C layout = row (v + 8*half), col (lane&15) ----
#pragma unroll
  for (int i = 0; i < 4; ++i) {
    int row_base = m0 + wm * 64 + i * 16;
#pragma unroll
    for (int j = 0; j < 4; ++j) {
      int col = n0 + wn * 64 + j * 16 + mlane;
      float bv = bias ? bias[col] : 0.0f;
#pragma unroll
      for (int v = 0; v < 8; ++v) {
        int row = row_base + v + 8 * half;
        store_out(&C[(size_t)row * N + col], acc[i][j][v] + bv);
      }
    }
  }
}

// ---------------------------------------------------------------------------
// Causal GQA flash attention (FA2-style, bf16 WMMA, fp32 softmax state).
// Block = 4 waves (128 thr); block owns (b, head, 64 q-rows); wave owns 16 rows.
// K tile staged in LDS [key][d]  == B-transposed layout for Q*K^T.
// V tile staged in LDS [d][key]  == B-transposed layout for P*V.
// P converted C-layout -> A-layout through a per-wave LDS buffer.
// ---------------------------------------------------------------------------
__global__ __launch_bounds__(128)
void gqa_attention_wmma(const unsigned short* __restrict__ Q,   // [B*S, 4096]
                        const unsigned short* __restrict__ Kc,  // [B*S, 1024]
                        const unsigned short* __restrict__ Vc,  // [B*S, 1024]
                        unsigned short* __restrict__ O,         // [B*S, 4096]
                        int S)
{
  __shared__ __align__(16) unsigned short lK[32][136];    // [key][d], pad 8
  __shared__ __align__(16) unsigned short lV[128][40];    // [d][key], pad 8
  __shared__ __align__(16) unsigned short lP[4][16][40];  // per-wave P tile

  const int tid   = threadIdx.x;
  const int lane  = tid & 31;
  const int wid   = tid >> 5;
  const int qtiles = S >> 6;

  int idx = blockIdx.x;
  const int qt = idx % qtiles; idx /= qtiles;
  const int hd = idx % 32;     idx /= 32;
  const int b  = idx;
  const int kvh = hd >> 2;                 // GROUP = 4 query heads per kv head
  const int q0  = qt * 64;
  const int qrow0 = q0 + wid * 16;

  const int mlane = lane & 15;
  const int half  = lane >> 4;

  // ---- Q fragments for this wave's 16 rows: 4 x (16x32) A-fragments ----
  Frag16 qf[4];
  {
    const unsigned short* qbase =
        Q + ((size_t)(b * S + qrow0 + mlane)) * 4096 + hd * 128;
#pragma unroll
    for (int kc = 0; kc < 4; ++kc) {
      qf[kc].u[0] = *(const v4u*)(qbase + kc * 32 + half * 8);
      qf[kc].u[1] = *(const v4u*)(qbase + kc * 32 + 16 + half * 8);
    }
  }

  const v8f vz = {0.f,0.f,0.f,0.f,0.f,0.f,0.f,0.f};
  v8f ot[8];                                  // O accum: 16 x 128 f32
#pragma unroll
  for (int t = 0; t < 8; ++t) ot[t] = vz;
  float mrun[8], lrun[8];
#pragma unroll
  for (int v = 0; v < 8; ++v) { mrun[v] = -1e30f; lrun[v] = 0.0f; }

  const float rs  = 0.08838834764831845f;     // 1/sqrt(128)
  const float L2E = 1.4426950408889634f;

  const int kend = q0 + 64;                   // causal horizon for this block
  for (int kt = 0; kt < kend; kt += 32) {
    // ---- cooperative K/V tile staging (32 keys x 128 dims) ----
#pragma unroll
    for (int i = 0; i < 4; ++i) {
      int v = tid + i * 128;
      int r = v >> 4, c = (v & 15) * 8;
      size_t src = ((size_t)(b * S + kt + r)) * 1024 + kvh * 128 + c;
      v4u kd = *(const v4u*)(Kc + src);
      *(v4u*)&lK[r][c] = kd;
      U4S8 vd; vd.u = *(const v4u*)(Vc + src);
#pragma unroll
      for (int e = 0; e < 8; ++e) lV[c + e][r] = vd.s[e];
    }
    // GL2 prefetch of next K/V tile (global_prefetch_b8)
    if (kt + 32 < kend) {
      size_t nsrc = ((size_t)(b * S + kt + 32 + (tid >> 2))) * 1024 + kvh * 128;
      __builtin_prefetch(Kc + nsrc, 0, 1);
      __builtin_prefetch(Vc + nsrc, 0, 1);
    }
    __syncthreads();

    // ---- S = Q * K^T : two 16x16 score tiles, 4 k32 WMMA steps each ----
    v8f sc[2]; sc[0] = vz; sc[1] = vz;
#pragma unroll
    for (int j = 0; j < 2; ++j)
#pragma unroll
      for (int kc = 0; kc < 4; ++kc) {
        Frag16 kf;
        const unsigned short* p = &lK[j * 16 + mlane][kc * 32 + half * 8];
        kf.u[0] = *(const v4u*)p;
        kf.u[1] = *(const v4u*)(p + 16);
        sc[j] = __builtin_amdgcn_wmma_f32_16x16x32_bf16(
            false, qf[kc].v, false, kf.v, (short)0, sc[j], false, false);
      }

    // ---- online softmax over the 32 new columns ----
#pragma unroll
    for (int v = 0; v < 8; ++v) {
      int row = qrow0 + v + 8 * half;
      int c0  = kt + mlane;
      int c1  = c0 + 16;
      float s0 = (c0 <= row) ? sc[0][v] * rs : -1e30f;
      float s1 = (c1 <= row) ? sc[1][v] * rs : -1e30f;
      float t = fmaxf(s0, s1);
#pragma unroll
      for (int d = 1; d < 16; d <<= 1) t = fmaxf(t, __shfl_xor(t, d, 32));
      float mnew  = fmaxf(mrun[v], t);
      float alpha = exp2f((mrun[v] - mnew) * L2E);
      float p0 = exp2f((s0 - mnew) * L2E);
      float p1 = exp2f((s1 - mnew) * L2E);
      float rsum = p0 + p1;
#pragma unroll
      for (int d = 1; d < 16; d <<= 1) rsum += __shfl_xor(rsum, d, 32);
      lrun[v] = lrun[v] * alpha + rsum;
      mrun[v] = mnew;
#pragma unroll
      for (int t8 = 0; t8 < 8; ++t8) ot[t8][v] *= alpha;   // rescale O rows
      // P tile to LDS (C-layout lanes -> row-major store)
      lP[wid][v + 8 * half][mlane]      = f2bf(p0);
      lP[wid][v + 8 * half][16 + mlane] = f2bf(p1);
    }

    // ---- O += P(16x32) * V(32x128) : 8 WMMA, reload P in A-layout ----
    Frag16 pf;
    {
      const unsigned short* p = &lP[wid][mlane][half * 8];
      pf.u[0] = *(const v4u*)p;
      pf.u[1] = *(const v4u*)(p + 16);
    }
#pragma unroll
    for (int t8 = 0; t8 < 8; ++t8) {
      Frag16 vf;
      const unsigned short* p = &lV[t8 * 16 + mlane][half * 8];
      vf.u[0] = *(const v4u*)p;
      vf.u[1] = *(const v4u*)(p + 16);
      ot[t8] = __builtin_amdgcn_wmma_f32_16x16x32_bf16(
          false, pf.v, false, vf.v, (short)0, ot[t8], false, false);
    }
    __syncthreads();
  }

  // ---- normalize and emit O as bf16 [B*S, H*128] ----
#pragma unroll
  for (int t8 = 0; t8 < 8; ++t8)
#pragma unroll
    for (int v = 0; v < 8; ++v) {
      int row = qrow0 + v + 8 * half;
      float val = ot[t8][v] / lrun[v];
      O[((size_t)(b * S + row)) * 4096 + hd * 128 + t8 * 16 + mlane] = f2bf(val);
    }
}

// ---------------------------------------------------------------------------
// Host-side orchestration
// ---------------------------------------------------------------------------
extern "C" void kernel_launch(void* const* d_in, const int* in_sizes, int n_in,
                              void* d_out, int out_size, void* d_ws, size_t ws_size,
                              hipStream_t stream) {
  (void)n_in; (void)out_size; (void)ws_size;
  const int D = 4096, KVD = 1024, S = 2048;

  const float* x  = (const float*)d_in[0];
  const float* Wq = (const float*)d_in[1];
  const float* bq = (const float*)d_in[2];
  const float* Wk = (const float*)d_in[3];
  const float* bk = (const float*)d_in[4];
  const float* Wv = (const float*)d_in[5];
  const float* bv = (const float*)d_in[6];
  const float* Wo = (const float*)d_in[7];
  const float* bo = (const float*)d_in[8];
  float* out = (float*)d_out;

  const int BS = in_sizes[0] / D;   // B*S = 4096
  const int Bn = BS / S;            // 2

  // workspace carve-out (bf16 copies + intermediates), ~192MB total
  char* ws = (char*)d_ws;
  size_t off = 0;
  auto carve = [&](size_t bytes) {
    char* p = ws + off;
    off += (bytes + 255) & ~(size_t)255;
    return (unsigned short*)p;
  };
  unsigned short* xb  = carve((size_t)BS * D * 2);
  unsigned short* Wqt = carve((size_t)D * D * 2);     // [N,K] transposed
  unsigned short* Wkt = carve((size_t)D * KVD * 2);
  unsigned short* Wvt = carve((size_t)D * KVD * 2);
  unsigned short* Wot = carve((size_t)D * D * 2);
  unsigned short* Qb  = carve((size_t)BS * D * 2);
  unsigned short* Kb  = carve((size_t)BS * KVD * 2);
  unsigned short* Vb  = carve((size_t)BS * KVD * 2);
  unsigned short* Ob  = carve((size_t)BS * D * 2);

  // x: plain convert
  {
    size_t n = (size_t)BS * D;
    f32_to_bf16_kernel<<<(int)((n + 255) / 256), 256, 0, stream>>>(x, xb, n);
  }
  // weights: convert + transpose to [N,K]
  auto convT = [&](const float* src, unsigned short* dst, int K, int N) {
    f32_to_bf16_transpose_kernel<<<dim3(N / 32, K / 32), 256, 0, stream>>>(src, dst, K, N);
  };
  convT(Wq, Wqt, D, D);
  convT(Wk, Wkt, D, KVD);
  convT(Wv, Wvt, D, KVD);
  convT(Wo, Wot, D, D);

  // Q/K/V projections (bf16 out)
  gemm_bf16_wmma<unsigned short>
      <<<dim3(D / 128, BS / 128), 128, 0, stream>>>(xb, Wqt, bq, Qb, BS, D, D);
  gemm_bf16_wmma<unsigned short>
      <<<dim3(KVD / 128, BS / 128), 128, 0, stream>>>(xb, Wkt, bk, Kb, BS, KVD, D);
  gemm_bf16_wmma<unsigned short>
      <<<dim3(KVD / 128, BS / 128), 128, 0, stream>>>(xb, Wvt, bv, Vb, BS, KVD, D);

  // causal GQA attention
  int attn_blocks = Bn * 32 * (S / 64);   // 2048
  gqa_attention_wmma<<<attn_blocks, 128, 0, stream>>>(Qb, Kb, Vb, Ob, S);

  // output projection (fp32 out)
  gemm_bf16_wmma<float>
      <<<dim3(D / 128, BS / 128), 128, 0, stream>>>(Ob, Wot, bo, out, BS, D, D);
}
